// MixerModel_67654324847149
// MI455X (gfx1250) — matrix operand order repaired and checked by reading, not compiled, not run on hardware
//
#include <hip/hip_runtime.h>
#include <math.h>

typedef __bf16 bf16;
typedef __attribute__((ext_vector_type(16))) __bf16 v16bf;
typedef __attribute__((ext_vector_type(8)))  __bf16 v8bf;
typedef __attribute__((ext_vector_type(8)))  float  v8f;

#define B_   4
#define N_   1024
#define M_   (B_ * N_)          // 4096 tokens
#define DM   1024               // d_model
#define DI   2048               // d_inner
#define DS   16                 // d_state
#define DTR  64                 // dt_rank
#define VGRID 33
#define NVOX (VGRID * VGRID * VGRID)   // 35937

static inline int cdiv(int a, int b) { return (a + b - 1) / b; }

// ---------------------------------------------------------------------------
// WMMA GEMM: C[M,N] = A[M,Kp](bf16) * B[Kp,Np](bf16) (+bias)(+act)
// act: 0 = none, 1 = relu, 3 = softplus
// Block tile 64x128, 4 waves (128 thr), each wave 32x64 via 8x
// v_wmma_f32_16x16x32_bf16 per K-step of 32.
// A staged row-major (frag runs contiguous -> ds_load_b128 x2);
// B staged fragment-major Bs[half][col][e] (-> ds_load_b128 x2, no packing).
// Kp multiple of 32, Np multiple of 128, M multiple of 64 (guaranteed by host).
// ---------------------------------------------------------------------------
__global__ __launch_bounds__(128) void k_gemm(
    const bf16* __restrict__ A, int Kp,
    const bf16* __restrict__ Bm, int Np,
    float* __restrict__ C, int N, int ldc,
    const float* __restrict__ bias, int act)
{
  __shared__ bf16 As[64][32];        // 4 KB
  __shared__ bf16 Bs[2][128][16];    // 8 KB, [K-half][col][K-elem]
  const int tid  = threadIdx.x;
  const int lane = tid & 31;
  const int wid  = tid >> 5;      // 0..3
  const int wm   = wid >> 1;      // 0..1 -> 32-row slab
  const int wn   = wid & 1;       // 0..1 -> 64-col slab
  const int rowBlk = blockIdx.y * 64;
  const int colBlk = blockIdx.x * 128;
  const int hf   = lane >> 4;     // lane half
  const int l15  = lane & 15;

  v8f acc[2][4] = {};

  for (int k0 = 0; k0 < Kp; k0 += 32) {
    // stage A 64x32 (coalesced: 32 consecutive cols per row)
    for (int i = tid; i < 64 * 32; i += 128) {
      int r = i >> 5, c = i & 31;
      As[r][c] = A[(size_t)(rowBlk + r) * Kp + (k0 + c)];
    }
    // stage B 32x128, fragment-major (coalesced global reads)
    for (int i = tid; i < 32 * 128; i += 128) {
      int r = i >> 7, c = i & 127;
      Bs[r >> 4][c][r & 15] = Bm[(size_t)(k0 + r) * Np + (colBlk + c)];
    }
    __syncthreads();

    v16bf af[2], bfr[4];
#pragma unroll
    for (int s = 0; s < 2; ++s) {
      int row = wm * 32 + s * 16 + l15;
      // A 16x32 layout (matches B): e=0..15 -> K=16*half+e
      v8bf lo = *(const v8bf*)&As[row][16 * hf];
      v8bf hi = *(const v8bf*)&As[row][16 * hf + 8];
      af[s] = __builtin_shufflevector(lo, hi, 0, 1, 2, 3, 4, 5, 6, 7,
                                      8, 9, 10, 11, 12, 13, 14, 15);
    }
#pragma unroll
    for (int t = 0; t < 4; ++t) {
      int col = wn * 64 + t * 16 + l15;
      // B 32x16 layout: lane col = lane&15, K = e + 16*half -> contiguous in Bs
      bfr[t] = *(const v16bf*)&Bs[hf][col][0];
    }
#pragma unroll
    for (int s = 0; s < 2; ++s)
#pragma unroll
      for (int t = 0; t < 4; ++t)
        acc[s][t] = __builtin_amdgcn_wmma_f32_16x16x32_bf16(
            false, af[s], false, bfr[t], (short)0, acc[s][t], false, false);
    __syncthreads();
  }

#pragma unroll
  for (int s = 0; s < 2; ++s)
#pragma unroll
    for (int t = 0; t < 4; ++t)
#pragma unroll
      for (int i = 0; i < 8; ++i) {
        int row = rowBlk + wm * 32 + s * 16 + 8 * hf + i;   // D layout: M = i + 8*half
        int col = colBlk + wn * 64 + t * 16 + l15;          // N = lane&15
        if (col < N) {
          float x = acc[s][t][i];
          if (bias) x += bias[col];
          if (act == 1) x = fmaxf(x, 0.f);
          else if (act == 3) x = (x > 20.f) ? x : log1pf(expf(x));
          C[(size_t)row * ldc + col] = x;
        }
      }
}

// ---------------------------------------------------------------------------
// conversions
// ---------------------------------------------------------------------------
__global__ void k_cvt_padK(bf16* __restrict__ dst, int Kp,
                           const float* __restrict__ src, int sld, int soff,
                           int K, int total)
{
  int i = blockIdx.x * 256 + threadIdx.x;
  if (i >= total) return;
  int r = i / Kp, c = i % Kp;
  dst[i] = (c < K) ? (bf16)src[(size_t)r * sld + soff + c] : (bf16)0.f;
}

__global__ void k_cvt_off(bf16* __restrict__ dst, int dld, int doff,
                          const float* __restrict__ src, int sld, int soff,
                          int cols, int total)
{
  int i = blockIdx.x * 256 + threadIdx.x;
  if (i >= total) return;
  int r = i / cols, c = i % cols;
  dst[(size_t)r * dld + doff + c] = (bf16)src[(size_t)r * sld + soff + c];
}

// W[N,K] row-major  ->  B[Kp,Np] bf16 (transposed, zero-padded)
__global__ void k_cvt_wT(bf16* __restrict__ dst, int Np,
                         const float* __restrict__ W, int N, int K, int total)
{
  int i = blockIdx.x * 256 + threadIdx.x;
  if (i >= total) return;
  int k = i / Np, n = i % Np;
  dst[i] = (k < K && n < N) ? (bf16)W[(size_t)n * K + k] : (bf16)0.f;
}

// ---------------------------------------------------------------------------
// elementwise helpers
// ---------------------------------------------------------------------------
__global__ void k_add_inplace(float* __restrict__ a, const float* __restrict__ b, int n)
{ int i = blockIdx.x * 256 + threadIdx.x; if (i < n) a[i] += b[i]; }

__global__ void k_copy(float* __restrict__ d, const float* __restrict__ s, int n)
{ int i = blockIdx.x * 256 + threadIdx.x; if (i < n) d[i] = s[i]; }

__global__ void k_zero_f(float* d, int n)
{ int i = blockIdx.x * 256 + threadIdx.x; if (i < n) d[i] = 0.f; }

__global__ void k_zero_u(unsigned* d, int n)
{ int i = blockIdx.x * 256 + threadIdx.x; if (i < n) d[i] = 0u; }

// ---------------------------------------------------------------------------
// normalizations
// ---------------------------------------------------------------------------
// LayerNorm per row with affine; out != in
__global__ __launch_bounds__(256) void k_ln(float* __restrict__ out,
                                            const float* __restrict__ in,
                                            const float* __restrict__ w,
                                            const float* __restrict__ b, int C)
{
  int row = blockIdx.x, tid = threadIdx.x;
  const float* x = in + (size_t)row * C;
  __shared__ float s1[256], s2[256];
  float a = 0.f, q = 0.f;
  for (int c = tid; c < C; c += 256) { float v = x[c]; a += v; q += v * v; }
  s1[tid] = a; s2[tid] = q; __syncthreads();
  for (int st = 128; st; st >>= 1) {
    if (tid < st) { s1[tid] += s1[tid + st]; s2[tid] += s2[tid + st]; }
    __syncthreads();
  }
  float m = s1[0] / C;
  float rstd = rsqrtf(s2[0] / C - m * m + 1e-5f);
  for (int c = tid; c < C; c += 256)
    out[(size_t)row * C + c] = (x[c] - m) * rstd * w[c] + b[c];
}

// InstanceNorm per row (no affine), optional relu, in-place. rows = gridDim.x
__global__ __launch_bounds__(256) void k_rownorm(float* __restrict__ X, int C, int relu)
{
  int row = blockIdx.x, tid = threadIdx.x;
  float* x = X + (size_t)row * C;
  __shared__ float s1[256], s2[256];
  float a = 0.f, q = 0.f;
  for (int c = tid; c < C; c += 256) { float v = x[c]; a += v; q += v * v; }
  s1[tid] = a; s2[tid] = q; __syncthreads();
  for (int st = 128; st; st >>= 1) {
    if (tid < st) { s1[tid] += s1[tid + st]; s2[tid] += s2[tid + st]; }
    __syncthreads();
  }
  float m = s1[0] / C;
  float rstd = rsqrtf(s2[0] / C - m * m + 1e-5f);
  for (int c = tid; c < C; c += 256) {
    float v = (x[c] - m) * rstd;
    x[c] = relu ? fmaxf(v, 0.f) : v;
  }
}

// InstanceNorm over points per (b,c) on X[(b*Nn+n)*C + c]; relu fused; in-place
__global__ __launch_bounds__(256) void k_inorm_bc(float* __restrict__ X, int C, int Nn)
{
  int b = blockIdx.x / C, c = blockIdx.x % C;
  int tid = threadIdx.x;
  size_t base = (size_t)b * Nn * C + c;
  __shared__ float s1[256], s2[256];
  float a = 0.f, q = 0.f;
  for (int n = tid; n < Nn; n += 256) {
    float v = X[base + (size_t)n * C]; a += v; q += v * v;
  }
  s1[tid] = a; s2[tid] = q; __syncthreads();
  for (int st = 128; st; st >>= 1) {
    if (tid < st) { s1[tid] += s1[tid + st]; s2[tid] += s2[tid + st]; }
    __syncthreads();
  }
  float m = s1[0] / Nn;
  float rstd = rsqrtf(s2[0] / Nn - m * m + 1e-5f);
  for (int n = tid; n < Nn; n += 256) {
    size_t p = base + (size_t)n * C;
    X[p] = fmaxf((X[p] - m) * rstd, 0.f);
  }
}

// BatchNorm over all R rows per channel c = blockIdx.x; affine + relu; in-place
__global__ __launch_bounds__(256) void k_bn(float* __restrict__ X,
                                            const float* __restrict__ g,
                                            const float* __restrict__ bb, int C, int R)
{
  int c = blockIdx.x, tid = threadIdx.x;
  __shared__ float s1[256], s2[256];
  float a = 0.f, q = 0.f;
  for (int r = tid; r < R; r += 256) {
    float v = X[(size_t)r * C + c]; a += v; q += v * v;
  }
  s1[tid] = a; s2[tid] = q; __syncthreads();
  for (int st = 128; st; st >>= 1) {
    if (tid < st) { s1[tid] += s1[tid + st]; s2[tid] += s2[tid + st]; }
    __syncthreads();
  }
  float m = s1[0] / R;
  float rstd = rsqrtf(s2[0] / R - m * m + 1e-5f);
  float gg = g[c], bv = bb[c];
  for (int r = tid; r < R; r += 256) {
    size_t p = (size_t)r * C + c;
    X[p] = fmaxf(gg * (X[p] - m) * rstd + bv, 0.f);
  }
}

// max over points per (b,c)
__global__ __launch_bounds__(256) void k_maxpool(float* __restrict__ out,
                                                 const float* __restrict__ X, int C, int Nn)
{
  int b = blockIdx.x / C, c = blockIdx.x % C;
  int tid = threadIdx.x;
  __shared__ float s1[256];
  float m = -3.4e38f;
  for (int n = tid; n < Nn; n += 256)
    m = fmaxf(m, X[((size_t)b * Nn + n) * C + c]);
  s1[tid] = m; __syncthreads();
  for (int st = 128; st; st >>= 1) {
    if (tid < st) s1[tid] = fmaxf(s1[tid], s1[tid + st]);
    __syncthreads();
  }
  if (tid == 0) out[(size_t)b * C + c] = s1[0];
}

// ---------------------------------------------------------------------------
// small FC (M=4) for the TNet head
// ---------------------------------------------------------------------------
__global__ void k_fc(float* __restrict__ out, const float* __restrict__ in,
                     const float* __restrict__ W, const float* __restrict__ bias,
                     int Bn, int O, int K)
{
  int i = blockIdx.x * 256 + threadIdx.x;
  if (i >= Bn * O) return;
  int b = i / O, o = i % O;
  float s = bias ? bias[o] : 0.f;
  const float* x = in + (size_t)b * K;
  const float* w = W + (size_t)o * K;
  for (int k = 0; k < K; ++k) s += x[k] * w[k];
  out[i] = s;
}

// apply 3x3 transform (+I): out[b,n,k] = sum_c pts[b,n,c] * (m9[b,c,k] + I)
__global__ void k_tapply(float* __restrict__ out, const float* __restrict__ pts,
                         const float* __restrict__ m9, int total)
{
  int i = blockIdx.x * 256 + threadIdx.x;
  if (i >= total) return;
  int b = i / N_;
  const float* Mv = m9 + (size_t)b * 9;
  float p0 = pts[(size_t)i * 3 + 0];
  float p1 = pts[(size_t)i * 3 + 1];
  float p2 = pts[(size_t)i * 3 + 2];
#pragma unroll
  for (int k = 0; k < 3; ++k) {
    float m0 = Mv[0 * 3 + k] + (k == 0 ? 1.f : 0.f);
    float m1 = Mv[1 * 3 + k] + (k == 1 ? 1.f : 0.f);
    float m2 = Mv[2 * 3 + k] + (k == 2 ? 1.f : 0.f);
    out[(size_t)i * 3 + k] = p0 * m0 + p1 * m1 + p2 * m2;
  }
}

// ---------------------------------------------------------------------------
// voxelization
// ---------------------------------------------------------------------------
__global__ __launch_bounds__(256) void k_minmax(float* __restrict__ mm,
                                                const float* __restrict__ p, int n)
{
  int tid = threadIdx.x;
  __shared__ float smn[3][256], smx[3][256];
  float mn[3] = {3.4e38f, 3.4e38f, 3.4e38f};
  float mx[3] = {-3.4e38f, -3.4e38f, -3.4e38f};
  for (int i = tid; i < n; i += 256) {
#pragma unroll
    for (int k = 0; k < 3; ++k) {
      float v = p[(size_t)i * 3 + k];
      mn[k] = fminf(mn[k], v); mx[k] = fmaxf(mx[k], v);
    }
  }
#pragma unroll
  for (int k = 0; k < 3; ++k) { smn[k][tid] = mn[k]; smx[k][tid] = mx[k]; }
  __syncthreads();
  for (int st = 128; st; st >>= 1) {
    if (tid < st)
#pragma unroll
      for (int k = 0; k < 3; ++k) {
        smn[k][tid] = fminf(smn[k][tid], smn[k][tid + st]);
        smx[k][tid] = fmaxf(smx[k][tid], smx[k][tid + st]);
      }
    __syncthreads();
  }
  if (tid < 3) { mm[tid] = smn[tid][0]; mm[3 + tid] = smx[tid][0]; }
}

__global__ void k_vox(float* __restrict__ X0, int* __restrict__ vid,
                      unsigned* __restrict__ cnt, const float* __restrict__ p,
                      const float* __restrict__ mm, int total)
{
  int i = blockIdx.x * 256 + threadIdx.x;
  if (i >= total) return;
  int b = i / N_;
  float v[3];
#pragma unroll
  for (int k = 0; k < 3; ++k) {
    float s = (mm[3 + k] - mm[k]) / (float)(VGRID - 1);   // (max-min)/H_GRID, H_GRID=32
    v[k] = floorf((p[(size_t)i * 3 + k] - mm[k]) / s);
    X0[(size_t)i * 4 + k] = v[k];
  }
  int id = (int)(v[0] * (float)(VGRID * VGRID) + v[1] * (float)VGRID + v[2]);
  vid[i] = id;
  atomicAdd(&cnt[(size_t)b * NVOX + id], 1u);
}

__global__ void k_gather(float* __restrict__ X0, const int* __restrict__ vid,
                         const unsigned* __restrict__ cnt, int total)
{
  int i = blockIdx.x * 256 + threadIdx.x;
  if (i >= total) return;
  int b = i / N_;
  X0[(size_t)i * 4 + 3] = (float)cnt[(size_t)b * NVOX + vid[i]];
}

// ---------------------------------------------------------------------------
// Mamba pieces
// ---------------------------------------------------------------------------
// depthwise causal conv (width 4) + SiLU; x lives in cols [0,DI) of xz (ld = 2*DI)
__global__ void k_dwconv(float* __restrict__ xt, const float* __restrict__ xz, int ldxz,
                         const float* __restrict__ w, const float* __restrict__ cb, int total)
{
  int i = blockIdx.x * 256 + threadIdx.x;
  if (i >= total) return;
  int d = i % DI;
  int t = (i / DI) % N_;
  int b = i / (DI * N_);
  float s = cb[d];
#pragma unroll
  for (int j = 0; j < 4; ++j) {
    int tt = t - 3 + j;
    if (tt >= 0) s += w[d * 4 + j] * xz[((size_t)(b * N_ + tt)) * ldxz + d];
  }
  float sg = 1.f / (1.f + expf(-s));
  xt[i] = s * sg;
}

// selective scan: one thread per (b,d) row, 16-state register scan over L
__global__ __launch_bounds__(256) void k_scan(
    float* __restrict__ y, const float* __restrict__ delta,
    const float* __restrict__ dbl, const float* __restrict__ xt,
    const float* __restrict__ xz, int ldxz, int zoff,
    const float* __restrict__ Alog, const float* __restrict__ Dp)
{
  int idx = blockIdx.x * 256 + threadIdx.x;
  if (idx >= B_ * DI) return;
  int d = idx % DI, b = idx / DI;
  float Ar[DS];
#pragma unroll
  for (int s = 0; s < DS; ++s) Ar[s] = -expf(Alog[(size_t)d * DS + s]);
  float h[DS] = {};
  float Dd = Dp[d];
  for (int t = 0; t < N_; ++t) {
    size_t r = (size_t)(b * N_ + t);
    float dl = delta[r * DI + d];
    float u  = xt[r * DI + d];
    float zv = xz[r * ldxz + zoff + d];
    float du = dl * u;
    const float* bc = dbl + r * 96;
    float yy = 0.f;
#pragma unroll
    for (int s = 0; s < DS; ++s) {
      h[s] = expf(dl * Ar[s]) * h[s] + du * bc[64 + s];
      yy += h[s] * bc[80 + s];
    }
    float sg = 1.f / (1.f + expf(-zv));
    y[r * DI + d] = (yy + u * Dd) * (zv * sg);
  }
}

// ---------------------------------------------------------------------------
// argmax over classes
// ---------------------------------------------------------------------------
__global__ void k_argmax(int* __restrict__ out, const float* __restrict__ lg,
                         int rows, int ncls, int ld)
{
  int r = blockIdx.x * 256 + threadIdx.x;
  if (r >= rows) return;
  const float* x = lg + (size_t)r * ld;
  float best = x[0]; int bi = 0;
  for (int c = 1; c < ncls; ++c) { float v = x[c]; if (v > best) { best = v; bi = c; } }
  out[r] = bi;
}

// ---------------------------------------------------------------------------
// host orchestration
// ---------------------------------------------------------------------------
extern "C" void kernel_launch(void* const* d_in, const int* in_sizes, int n_in,
                              void* d_out, int out_size, void* d_ws, size_t ws_size,
                              hipStream_t stream)
{
  (void)in_sizes; (void)n_in; (void)out_size; (void)ws_size;
  const float* pts    = (const float*)d_in[0];
  const float* tn_c1w = (const float*)d_in[1],  *tn_c1b = (const float*)d_in[2];
  const float* tn_c2w = (const float*)d_in[3],  *tn_c2b = (const float*)d_in[4];
  const float* tn_c3w = (const float*)d_in[5],  *tn_c3b = (const float*)d_in[6];
  const float* tn_f1w = (const float*)d_in[7],  *tn_f1b = (const float*)d_in[8];
  const float* tn_f2w = (const float*)d_in[9],  *tn_f2b = (const float*)d_in[10];
  const float* tn_f3w = (const float*)d_in[11], *tn_f3b = (const float*)d_in[12];
  const float* em_c1w = (const float*)d_in[13], *em_c1b = (const float*)d_in[14];
  const float* em_g1  = (const float*)d_in[15], *em_b1  = (const float*)d_in[16];
  const float* em_c2w = (const float*)d_in[17], *em_c2b = (const float*)d_in[18];
  const float* em_g2  = (const float*)d_in[19], *em_b2  = (const float*)d_in[20];
  const float* em_c3w = (const float*)d_in[21], *em_c3b = (const float*)d_in[22];
  const float* em_g3  = (const float*)d_in[23], *em_b3  = (const float*)d_in[24];
  const float* ly_nw  = (const float*)d_in[25], *ly_nb  = (const float*)d_in[26];
  const float* ly_inw = (const float*)d_in[27];
  const float* ly_cw  = (const float*)d_in[28], *ly_cb  = (const float*)d_in[29];
  const float* ly_xw  = (const float*)d_in[30];
  const float* ly_dtw = (const float*)d_in[31], *ly_dtb = (const float*)d_in[32];
  const float* ly_Al  = (const float*)d_in[33], *ly_Dp  = (const float*)d_in[34];
  const float* ly_ow  = (const float*)d_in[35];
  const float* nfw    = (const float*)d_in[36], *nfb    = (const float*)d_in[37];
  const float* mw1 = (const float*)d_in[38], *mb1 = (const float*)d_in[39];
  const float* mw2 = (const float*)d_in[40], *mb2 = (const float*)d_in[41];
  const float* mw3 = (const float*)d_in[42], *mb3 = (const float*)d_in[43];
  const float* mw4 = (const float*)d_in[44], *mb4 = (const float*)d_in[45];
  const float* mw5 = (const float*)d_in[46], *mb5 = (const float*)d_in[47];
  int* out = (int*)d_out;

  // --- workspace allocation ---
  char* ws = (char*)d_ws; size_t off = 0;
  auto alloc = [&](size_t bytes) -> void* {
    off = (off + 255) & ~(size_t)255;
    void* p = ws + off; off += bytes; return p;
  };
  float* XZ    = (float*)alloc((size_t)M_ * 2 * DI * 4);
  float* XT    = (float*)alloc((size_t)M_ * DI * 4);
  float* DELTA = (float*)alloc((size_t)M_ * DI * 4);
  float* Yb    = (float*)alloc((size_t)M_ * DI * 4);
  float* H     = (float*)alloc((size_t)M_ * DM * 4);
  float* RES   = (float*)alloc((size_t)M_ * DM * 4);
  float* HN    = (float*)alloc((size_t)M_ * DM * 4);
  float* SKIP  = (float*)alloc((size_t)M_ * DM * 4);
  float* T64   = (float*)alloc((size_t)M_ * 64 * 4);
  float* T128  = (float*)alloc((size_t)M_ * 128 * 4);
  float* DBL   = (float*)alloc((size_t)M_ * 96 * 4);
  float* POOL  = (float*)alloc((size_t)B_ * 1024 * 4);
  float* F1    = (float*)alloc((size_t)B_ * 512 * 4);
  float* F2    = (float*)alloc((size_t)B_ * 256 * 4);
  float* F3    = (float*)alloc((size_t)B_ * 16 * 4);
  float* PTS2  = (float*)alloc((size_t)M_ * 3 * 4);
  float* X0    = (float*)alloc((size_t)M_ * 4 * 4);
  float* MMx   = (float*)alloc(8 * 4);
  float* LOG   = (float*)alloc((size_t)M_ * 40 * 4);
  int*      VID = (int*)alloc((size_t)M_ * 4);
  unsigned* CNT = (unsigned*)alloc((size_t)B_ * NVOX * 4);
  bf16* ABF = (bf16*)alloc((size_t)M_ * DI * 2);        // activations, <= 4096x2048
  bf16* WBF = (bf16*)alloc((size_t)(2 * DI) * DM * 2);  // weights^T, <= 1024x4096
  float* T1024 = XZ;   // aliases (lifetimes disjoint)
  float* Ml1 = XT; float* Ml2 = DELTA; float* Ml3 = Yb; float* Ml4 = XZ;

  auto cvtA = [&](const float* src, int sld, int soff, int K, int Kp) {
    int total = M_ * Kp;
    k_cvt_padK<<<cdiv(total, 256), 256, 0, stream>>>(ABF, Kp, src, sld, soff, K, total);
  };
  auto cvtW = [&](const float* W, int N, int K, int Kp, int Np) {
    int total = Kp * Np;
    k_cvt_wT<<<cdiv(total, 256), 256, 0, stream>>>(WBF, Np, W, N, K, total);
  };
  auto gemm = [&](int Kp, int Np, float* C, int N, int ldc, const float* bias, int act) {
    dim3 g(Np / 128, M_ / 64);
    k_gemm<<<g, 128, 0, stream>>>(ABF, Kp, WBF, Np, C, N, ldc, bias, act);
  };

  // ================= TNet =================
  cvtA(pts, 3, 0, 3, 32);            cvtW(tn_c1w, 64, 3, 32, 128);
  gemm(32, 128, T64, 64, 64, tn_c1b, 0);
  k_inorm_bc<<<B_ * 64, 256, 0, stream>>>(T64, 64, N_);
  cvtA(T64, 64, 0, 64, 64);          cvtW(tn_c2w, 128, 64, 64, 128);
  gemm(64, 128, T128, 128, 128, tn_c2b, 0);
  k_inorm_bc<<<B_ * 128, 256, 0, stream>>>(T128, 128, N_);
  cvtA(T128, 128, 0, 128, 128);      cvtW(tn_c3w, 1024, 128, 128, 1024);
  gemm(128, 1024, T1024, 1024, 1024, tn_c3b, 0);
  k_inorm_bc<<<B_ * 1024, 256, 0, stream>>>(T1024, 1024, N_);
  k_maxpool<<<B_ * 1024, 256, 0, stream>>>(POOL, T1024, 1024, N_);
  k_fc<<<cdiv(B_ * 512, 256), 256, 0, stream>>>(F1, POOL, tn_f1w, tn_f1b, B_, 512, 1024);
  k_rownorm<<<B_, 256, 0, stream>>>(F1, 512, 1);
  k_fc<<<cdiv(B_ * 256, 256), 256, 0, stream>>>(F2, F1, tn_f2w, tn_f2b, B_, 256, 512);
  k_rownorm<<<B_, 256, 0, stream>>>(F2, 256, 1);
  k_fc<<<1, 256, 0, stream>>>(F3, F2, tn_f3w, tn_f3b, B_, 9, 256);
  k_tapply<<<cdiv(M_, 256), 256, 0, stream>>>(PTS2, pts, F3, M_);

  // ================= voxel embed =================
  k_minmax<<<1, 256, 0, stream>>>(MMx, PTS2, M_);
  k_zero_u<<<cdiv(B_ * NVOX, 256), 256, 0, stream>>>(CNT, B_ * NVOX);
  k_vox<<<cdiv(M_, 256), 256, 0, stream>>>(X0, VID, CNT, PTS2, MMx, M_);
  k_gather<<<cdiv(M_, 256), 256, 0, stream>>>(X0, VID, CNT, M_);

  // ================= point-cloud embedding =================
  cvtA(X0, 4, 0, 4, 32);             cvtW(em_c1w, 64, 4, 32, 128);
  gemm(32, 128, T64, 64, 64, em_c1b, 0);
  k_bn<<<64, 256, 0, stream>>>(T64, em_g1, em_b1, 64, M_);
  cvtA(T64, 64, 0, 64, 64);          cvtW(em_c2w, 128, 64, 64, 128);
  gemm(64, 128, T128, 128, 128, em_c2b, 0);
  k_bn<<<128, 256, 0, stream>>>(T128, em_g2, em_b2, 128, M_);
  cvtA(T128, 128, 0, 128, 128);      cvtW(em_c3w, 1024, 128, 128, 1024);
  gemm(128, 1024, H, 1024, 1024, em_c3b, 0);
  k_bn<<<1024, 256, 0, stream>>>(H, em_g3, em_b3, 1024, M_);
  k_copy<<<cdiv(M_ * DM, 256), 256, 0, stream>>>(SKIP, H, M_ * DM);
  k_zero_f<<<cdiv(M_ * DM, 256), 256, 0, stream>>>(RES, M_ * DM);

  // ================= Mamba layers =================
  for (int l = 0; l < 4; ++l) {
    k_add_inplace<<<cdiv(M_ * DM, 256), 256, 0, stream>>>(RES, H, M_ * DM);
    k_ln<<<M_, 256, 0, stream>>>(HN, RES, ly_nw + (size_t)l * DM, ly_nb + (size_t)l * DM, DM);
    // in_proj: [4096,1024] x [1024,4096]
    cvtA(HN, DM, 0, DM, DM);
    cvtW(ly_inw + (size_t)l * 2 * DI * DM, 2 * DI, DM, DM, 2 * DI);
    gemm(DM, 2 * DI, XZ, 2 * DI, 2 * DI, nullptr, 0);
    // depthwise causal conv + silu
    k_dwconv<<<cdiv(M_ * DI, 256), 256, 0, stream>>>(XT, XZ, 2 * DI,
        ly_cw + (size_t)l * DI * 4, ly_cb + (size_t)l * DI, M_ * DI);
    // x_proj: [4096,2048] x [2048,96]
    cvtA(XT, DI, 0, DI, DI);
    cvtW(ly_xw + (size_t)l * 96 * DI, 96, DI, DI, 128);
    gemm(DI, 128, DBL, 96, 96, nullptr, 0);
    // dt_proj + softplus: [4096,64] x [64,2048]
    cvtA(DBL, 96, 0, DTR, DTR);
    cvtW(ly_dtw + (size_t)l * DI * DTR, DI, DTR, DTR, DI);
    gemm(DTR, DI, DELTA, DI, DI, ly_dtb + (size_t)l * DI, 3);
    // selective scan (+ D skip, + silu(z) gate)
    k_scan<<<cdiv(B_ * DI, 256), 256, 0, stream>>>(Yb, DELTA, DBL, XT, XZ, 2 * DI, DI,
        ly_Al + (size_t)l * DI * DS, ly_Dp + (size_t)l * DI);
    // out_proj: [4096,2048] x [2048,1024]
    cvtA(Yb, DI, 0, DI, DI);
    cvtW(ly_ow + (size_t)l * DM * DI, DM, DI, DI, DM);
    gemm(DI, DM, H, DM, DM, nullptr, 0);
  }

  // ================= head =================
  k_add_inplace<<<cdiv(M_ * DM, 256), 256, 0, stream>>>(RES, H, M_ * DM);
  k_ln<<<M_, 256, 0, stream>>>(HN, RES, nfw, nfb, DM);
  {
    int total = M_ * DM;  // concat [HN | SKIP] into ABF [4096,2048]
    k_cvt_off<<<cdiv(total, 256), 256, 0, stream>>>(ABF, 2048, 0,    HN,   DM, 0, DM, total);
    k_cvt_off<<<cdiv(total, 256), 256, 0, stream>>>(ABF, 2048, 1024, SKIP, DM, 0, DM, total);
  }
  cvtW(mw1, 1024, 2048, 2048, 1024);
  gemm(2048, 1024, Ml1, 1024, 1024, mb1, 1);
  cvtA(Ml1, 1024, 0, 1024, 1024); cvtW(mw2, 512, 1024, 1024, 512);
  gemm(1024, 512, Ml2, 512, 512, mb2, 1);
  cvtA(Ml2, 512, 0, 512, 512);    cvtW(mw3, 256, 512, 512, 256);
  gemm(512, 256, Ml3, 256, 256, mb3, 1);
  cvtA(Ml3, 256, 0, 256, 256);    cvtW(mw4, 128, 256, 256, 128);
  gemm(256, 128, Ml4, 128, 128, mb4, 1);
  cvtA(Ml4, 128, 0, 128, 128);    cvtW(mw5, 40, 128, 128, 128);
  gemm(128, 128, LOG, 40, 40, mb5, 0);
  k_argmax<<<cdiv(M_, 256), 256, 0, stream>>>(out, LOG, M_, 40, 40);
}
